// Experts_53566832115811
// MI455X (gfx1250) — compile-verified, matrix operand order checked by reading, not run on hardware
//
#include <hip/hip_runtime.h>
#include <hip/hip_bf16.h>
#include <math.h>

// Problem sizes (fixed by the reference)
#define Bq    4
#define Eq    8
#define Nq    512
#define Dq    1024
#define DFFq  4096
#define NAq   256

typedef __attribute__((ext_vector_type(16))) __bf16 v16bf;
typedef __attribute__((ext_vector_type(8)))  float  v8f;

union FragBF {
    v16bf bf;
    uint4 u[2];
};

__device__ __forceinline__ unsigned short f2bf(float f) {
    unsigned u = __float_as_uint(f);
    u += 0x7FFFu + ((u >> 16) & 1u);   // round-to-nearest-even
    return (unsigned short)(u >> 16);
}

__device__ __forceinline__ float gelu_tanh(float x) {
    const float k0 = 0.7978845608028654f;   // sqrt(2/pi)
    const float k1 = 0.044715f;
    float u = k0 * (x + k1 * x * x * x);
#if __has_builtin(__builtin_amdgcn_tanhf)
    float t = __builtin_amdgcn_tanhf(u);    // CDNA5 v_tanh_f32
#else
    float ex = __expf(2.0f * u);            // branch-free tanh
    float t  = 1.0f - __fdividef(2.0f, ex + 1.0f);
#endif
    return 0.5f * x * (1.0f + t);
}

// ---- 16-byte global -> LDS copy: async-to-LDS if available ------------------
typedef int v4i_vs __attribute__((vector_size(16)));

#if __has_builtin(__builtin_amdgcn_global_load_async_to_lds_b128)
#define ASYNC_LDS 1
__device__ __forceinline__ void copy16(const void* g, void* l) {
    __builtin_amdgcn_global_load_async_to_lds_b128(
        (__attribute__((address_space(1))) v4i_vs*)(g),
        (__attribute__((address_space(3))) v4i_vs*)(l), 0, 0);
}
__device__ __forceinline__ void wait_async() {
    asm volatile("s_wait_asynccnt 0" ::: "memory");
}
#else
#define ASYNC_LDS 0
__device__ __forceinline__ void copy16(const void* g, void* l) {
    *(uint4*)l = *(const uint4*)g;
}
__device__ __forceinline__ void wait_async() {}
#endif

// ---- kernel 0: audio_pooled[b][d] = mean_na audio[b][na][d] ----------------
__global__ void pool_kernel(const float* __restrict__ audio,
                            float* __restrict__ pooled) {
    int idx = blockIdx.x * blockDim.x + threadIdx.x;     // B*D
    if (idx >= Bq * Dq) return;
    int b = idx / Dq, d = idx % Dq;
    const float* p = audio + (size_t)b * NAq * Dq + d;
    float s = 0.f;
    for (int i = 0; i < NAq; ++i) s += p[(size_t)i * Dq];
    pooled[idx] = s * (1.0f / NAq);
}

// ---- kernel 1: bias[e][b][f] = b1[e][f] + sum_d pooled[b][d]*Wa[e][d][f] ---
__global__ void bias_kernel(const float* __restrict__ pooled,
                            const float* __restrict__ Wa,
                            const float* __restrict__ b1,
                            float* __restrict__ bias) {
    int idx = blockIdx.x * blockDim.x + threadIdx.x;     // E*DFF
    if (idx >= Eq * DFFq) return;
    int e = idx / DFFq, f = idx % DFFq;
    const float* wac = Wa + (size_t)e * Dq * DFFq + f;
    float a0 = 0.f, a1 = 0.f, a2 = 0.f, a3 = 0.f;
    for (int d = 0; d < Dq; ++d) {
        float w = wac[(size_t)d * DFFq];
        a0 += pooled[0 * Dq + d] * w;
        a1 += pooled[1 * Dq + d] * w;
        a2 += pooled[2 * Dq + d] * w;
        a3 += pooled[3 * Dq + d] * w;
    }
    float bb = b1[idx];
    bias[((size_t)e * Bq + 0) * DFFq + f] = a0 + bb;
    bias[((size_t)e * Bq + 1) * DFFq + f] = a1 + bb;
    bias[((size_t)e * Bq + 2) * DFFq + f] = a2 + bb;
    bias[((size_t)e * Bq + 3) * DFFq + f] = a3 + bb;
}

// ---- kernel 2: per-expert f32 [R][C] -> bf16 transposed [C][R] -------------
// grid: (C/32, R/32, E), 256 threads
__global__ void transpose_bf16_kernel(const float* __restrict__ src,
                                      unsigned short* __restrict__ dst,
                                      int R, int C) {
    __shared__ unsigned short tile[32][33];
    int e  = blockIdx.z;
    int r0 = blockIdx.y * 32;
    int c0 = blockIdx.x * 32;
    const float* s = src + (size_t)e * R * C;
    unsigned short* t = dst + (size_t)e * R * C;

    int tr = threadIdx.x >> 3;        // 0..31
    int tc = (threadIdx.x & 7) * 4;   // 0..28
    float4 v = *(const float4*)(s + (size_t)(r0 + tr) * C + c0 + tc);
    tile[tc + 0][tr] = f2bf(v.x);
    tile[tc + 1][tr] = f2bf(v.y);
    tile[tc + 2][tr] = f2bf(v.z);
    tile[tc + 3][tr] = f2bf(v.w);
    __syncthreads();
    // tile[cl][rl] = src(r0+rl, c0+cl); write dst row (c0+tr), cols r0+tc..+3
    unsigned a0 = tile[tr][tc + 0], a1 = tile[tr][tc + 1];
    unsigned a2 = tile[tr][tc + 2], a3 = tile[tr][tc + 3];
    uint2 w;
    w.x = a0 | (a1 << 16);
    w.y = a2 | (a3 << 16);
    *(uint2*)(t + (size_t)(c0 + tr) * R + r0 + tc) = w;
}

// ---- main fused kernel: per (e,b,ntile) compute 32x1024 output tile --------
// 8 waves (wave32). GEMM1: 8 waves each own one 16x16 tile of H[32,64] chunk.
// GEMM2: wave owns 16 rows x 256 cols of output (16 wmma acc tiles, 128 VGPRs).
__global__ __launch_bounds__(256, 1)
void moe_ffn_kernel(const float* __restrict__ x,
                    const unsigned short* __restrict__ W1T,   // [E][DFF][D] bf16
                    const unsigned short* __restrict__ W2T,   // [E][D][DFF] bf16
                    const float* __restrict__ b2,
                    const float* __restrict__ bias,
                    float* __restrict__ out) {
    __shared__ unsigned short Xs[32][1032];     //  66,048 B  X tile (bf16, row-major)
    __shared__ unsigned short W1s[64][1032];    // 132,096 B  W1 panel  [f_local][k]
    __shared__ unsigned short W2s[1024][40];    //  81,920 B  W2 half   [d][f_local]
    __shared__ unsigned short Hs[32][72];       //   4,608 B  H chunk (bf16)
                                                // total ~278 KB of 320 KB WGP LDS

    const int tid  = threadIdx.x;
    const int wave = tid >> 5;
    const int lane = tid & 31;
    const int nt   = blockIdx.x;   // 0..15
    const int b    = blockIdx.y;   // 0..3
    const int e    = blockIdx.z;   // 0..7

    const unsigned short* W1Te = W1T + (size_t)e * DFFq * Dq;
    const unsigned short* W2Te = W2T + (size_t)e * Dq * DFFq;
    const float* b2e    = b2 + (size_t)e * Dq;
    const float* biasEB = bias + ((size_t)e * Bq + b) * DFFq;

    // ---- stage X tile: 32x1024 f32 -> bf16 LDS (once) ----------------------
    {
        int r  = tid >> 3;
        int c0 = (tid & 7) * 128;
        const float* src =
            x + (((size_t)b * Eq + e) * Nq + (size_t)nt * 32 + r) * Dq + c0;
#pragma unroll 8
        for (int j = 0; j < 128; j += 4) {
            float4 v = *(const float4*)(src + j);
            Xs[r][c0 + j + 0] = f2bf(v.x);
            Xs[r][c0 + j + 1] = f2bf(v.y);
            Xs[r][c0 + j + 2] = f2bf(v.z);
            Xs[r][c0 + j + 3] = f2bf(v.w);
        }
    }

    const int mi  = wave >> 2;            // GEMM1 row tile (0..1)
    const int fi  = wave & 3;             // GEMM1 col tile (0..3)
    const int l16 = lane & 15;
    const int kh  = (lane >> 4) * 16;     // lane-half K offset (consistent A/B packing)

    const int r2 = (wave & 1) * 16;       // GEMM2 row group
    const int c2 = (wave >> 1) * 256;     // GEMM2 col group

    const v8f vzero = {0.f, 0.f, 0.f, 0.f, 0.f, 0.f, 0.f, 0.f};
    v8f acc[16];
#pragma unroll
    for (int i = 0; i < 16; ++i) acc[i] = vzero;

    for (int f0 = 0; f0 < DFFq; f0 += 64) {
        __syncthreads();   // previous chunk fully consumed

        // ---- stage W1 panel: W1s[fl][k] <- W1Te[f0+fl][k]  (64 x 2 KB) ----
        {
#pragma unroll 4
            for (int i = 0; i < 32; ++i) {
                int id  = i * 256 + tid;       // 8192 chunks of 16 B
                int row = id >> 7;             // 0..63
                int c   = (id & 127) * 8;      // ushort offset
                copy16(W1Te + (size_t)(f0 + row) * Dq + c, &W1s[row][c]);
            }
        }
        // ---- stage W2 half 0: W2s[d][fl] <- W2Te[d][f0+fl], fl<32 ----------
        {
#pragma unroll 4
            for (int i = 0; i < 16; ++i) {
                int id = i * 256 + tid;        // 4096 chunks of 16 B
                int d  = id >> 2;
                int c  = (id & 3) * 8;
                copy16(W2Te + (size_t)d * DFFq + f0 + c, &W2s[d][c]);
            }
        }
        wait_async();
        __syncthreads();

        // ---- GEMM1: one 16x16 tile of H per wave, K=1024 -------------------
        {
            v8f h = vzero;
            const int arow = mi * 16 + l16;
            const int brow = fi * 16 + l16;
#pragma unroll 4
            for (int k = 0; k < Dq; k += 32) {
                FragBF a, w;
                a.u[0] = *(const uint4*)&Xs[arow][k + kh];
                a.u[1] = *(const uint4*)&Xs[arow][k + kh + 8];
                w.u[0] = *(const uint4*)&W1s[brow][k + kh];
                w.u[1] = *(const uint4*)&W1s[brow][k + kh + 8];
                h = __builtin_amdgcn_wmma_f32_16x16x32_bf16(
                        false, a.bf, false, w.bf, (short)0, h, false, false);
            }
            float bv  = biasEB[f0 + fi * 16 + l16];
            int   hm0 = mi * 16 + (lane >> 4) * 8;
            int   hn  = fi * 16 + l16;
#pragma unroll
            for (int v = 0; v < 8; ++v)
                Hs[hm0 + v][hn] = f2bf(gelu_tanh(h[v] + bv));
        }
        __syncthreads();

        // ---- GEMM2 accumulate, K-half 0 (f0 .. f0+31) ----------------------
        {
            FragBF a;
            a.u[0] = *(const uint4*)&Hs[r2 + l16][kh];
            a.u[1] = *(const uint4*)&Hs[r2 + l16][kh + 8];
#pragma unroll
            for (int ci = 0; ci < 16; ++ci) {
                FragBF w;
                int d = c2 + ci * 16 + l16;
                w.u[0] = *(const uint4*)&W2s[d][kh];
                w.u[1] = *(const uint4*)&W2s[d][kh + 8];
                acc[ci] = __builtin_amdgcn_wmma_f32_16x16x32_bf16(
                              false, a.bf, false, w.bf, (short)0, acc[ci],
                              false, false);
            }
        }
        __syncthreads();   // W2s half 0 consumed

        // ---- stage W2 half 1 ----------------------------------------------
        {
#pragma unroll 4
            for (int i = 0; i < 16; ++i) {
                int id = i * 256 + tid;
                int d  = id >> 2;
                int c  = (id & 3) * 8;
                copy16(W2Te + (size_t)d * DFFq + f0 + 32 + c, &W2s[d][c]);
            }
        }
        wait_async();
        __syncthreads();

        // ---- GEMM2 accumulate, K-half 1 (f0+32 .. f0+63) -------------------
        {
            FragBF a;
            a.u[0] = *(const uint4*)&Hs[r2 + l16][32 + kh];
            a.u[1] = *(const uint4*)&Hs[r2 + l16][32 + kh + 8];
#pragma unroll
            for (int ci = 0; ci < 16; ++ci) {
                FragBF w;
                int d = c2 + ci * 16 + l16;
                w.u[0] = *(const uint4*)&W2s[d][kh];
                w.u[1] = *(const uint4*)&W2s[d][kh + 8];
                acc[ci] = __builtin_amdgcn_wmma_f32_16x16x32_bf16(
                              false, a.bf, false, w.bf, (short)0, acc[ci],
                              false, false);
            }
        }
    }

    // ---- epilogue: out = acc + b2 -----------------------------------------
    {
        int m0 = (lane >> 4) * 8;
        size_t rowbase =
            (((size_t)b * Eq + e) * Nq + (size_t)nt * 32 + r2 + m0) * Dq;
#pragma unroll
        for (int ci = 0; ci < 16; ++ci) {
            int d = c2 + ci * 16 + l16;
            float bb = b2e[d];
#pragma unroll
            for (int v = 0; v < 8; ++v)
                out[rowbase + (size_t)v * Dq + d] = acc[ci][v] + bb;
        }
    }
}

extern "C" void kernel_launch(void* const* d_in, const int* in_sizes, int n_in,
                              void* d_out, int out_size, void* d_ws, size_t ws_size,
                              hipStream_t stream) {
    const float* x     = (const float*)d_in[0];
    const float* audio = (const float*)d_in[1];
    const float* W1    = (const float*)d_in[2];
    const float* b1    = (const float*)d_in[3];
    const float* Wa    = (const float*)d_in[4];
    const float* W2    = (const float*)d_in[5];
    const float* b2    = (const float*)d_in[6];
    float* out = (float*)d_out;

    // workspace layout
    float* pooled = (float*)d_ws;                               // 16 KB
    float* bias   = pooled + (size_t)Bq * Dq;                   // 512 KB
    unsigned short* W1T = (unsigned short*)((char*)d_ws + (1u << 20));   // 64 MiB
    unsigned short* W2T = W1T + (size_t)Eq * DFFq * Dq;                  // 64 MiB

    pool_kernel<<<(Bq * Dq + 255) / 256, 256, 0, stream>>>(audio, pooled);
    bias_kernel<<<(Eq * DFFq + 255) / 256, 256, 0, stream>>>(pooled, Wa, b1, bias);

    // W1 [E][D][DFF] -> W1T [E][DFF][D] bf16 ; W2 [E][DFF][D] -> W2T [E][D][DFF]
    transpose_bf16_kernel<<<dim3(DFFq / 32, Dq / 32, Eq), 256, 0, stream>>>(
        W1, W1T, Dq, DFFq);
    transpose_bf16_kernel<<<dim3(Dq / 32, DFFq / 32, Eq), 256, 0, stream>>>(
        W2, W2T, DFFq, Dq);

    dim3 grid(Nq / 32, Bq, Eq);   // 16 n-tiles x 4 batch x 8 experts = 512 WGs
    moe_ffn_kernel<<<grid, 256, 0, stream>>>(x, W1T, W2T, b2, bias, out);
}
